// HierarchicalChronoFormer_23802708754627
// MI455X (gfx1250) — compile-verified, matrix operand order checked by reading, not run on hardware
//
#include <hip/hip_runtime.h>
#include <hip/hip_bf16.h>

// ---------------------------------------------------------------------------
// HierarchicalChronoFormer for MI455X (gfx1250, wave32, WMMA bf16 16x16x32)
// B=8, S=4096, D=512, H=8, dk=64, BIN=64, NB=64, T_BINS=1000
//
// bf16 activations + v_wmma_f32_16x16x32_bf16 (f32 accumulate); async
// global->LDS copies (ASYNCcnt) for WMMA operand staging; LDS tiles laid out
// so every fragment read is K-contiguous (ds_load_b128).
// ---------------------------------------------------------------------------

#define USE_ASYNC_COPY 1

typedef __bf16 v16bf __attribute__((ext_vector_type(16)));
typedef float  v8f   __attribute__((ext_vector_type(8)));

#define N_HEADS 8
#define DIMC    512
#define TBINS   1000

__device__ __forceinline__ v8f wmma_bf16(v16bf a, v16bf b, v8f c) {
  return __builtin_amdgcn_wmma_f32_16x16x32_bf16(false, a, false, b,
                                                 (short)0, c, false, false);
}

// 16-bit A-matrix 16x32 lane/K map (CDNA5 ISA §7.12.2):
// lane<16: K = {0..7} in V0..3, {16..23} in V4..7 ; lane>=16: +8 shift.
__device__ __forceinline__ int kmap(int j, int lane) {
  int kb = (lane < 16) ? 0 : 8;
  return (j < 8) ? (kb + j) : (16 + kb + (j - 8));
}

// Async 16B global->LDS copy. Generic LDS pointers are {SHARED_BASE, offset}
// (ISA §10.2 aperture rules), so the low 32 bits are the LDS byte address the
// VDST operand wants. Tracked by ASYNCcnt; pair with s_wait_asynccnt.
__device__ __forceinline__ void async_copy_b128(void* lds_dst, const void* gsrc) {
#if USE_ASYNC_COPY
  unsigned lds = (unsigned)(uintptr_t)lds_dst;
  unsigned long long ga = (unsigned long long)(uintptr_t)gsrc;
  asm volatile("global_load_async_to_lds_b128 %0, %1, off"
               :: "v"(lds), "v"(ga) : "memory");
#else
  *reinterpret_cast<uint4*>(lds_dst) = *reinterpret_cast<const uint4*>(gsrc);
#endif
}

__device__ __forceinline__ void async_copy_fence() {
#if USE_ASYNC_COPY
  asm volatile("s_wait_asynccnt 0x0" ::: "memory");
#endif
}

// ---------------------------------------------------------------------------
// 1) per-batch inclusive cumsum of max(td,0)  (S=4096, one block per batch)
// ---------------------------------------------------------------------------
__global__ __launch_bounds__(256) void cumsum_kernel(
    const float* __restrict__ td, float* __restrict__ abs_t, int S) {
  __shared__ float sc[256];
  __shared__ float carry;
  const int b = blockIdx.x, t = threadIdx.x;
  if (t == 0) carry = 0.0f;
  __syncthreads();
  for (int c0 = 0; c0 < S; c0 += 256) {
    float v = td[(size_t)b * S + c0 + t];
    v = fmaxf(v, 0.0f);
    sc[t] = v;
    __syncthreads();
    for (int off = 1; off < 256; off <<= 1) {       // Hillis-Steele scan
      float add = (t >= off) ? sc[t - off] : 0.0f;
      __syncthreads();
      sc[t] += add;
      __syncthreads();
    }
    float prev = carry;
    abs_t[(size_t)b * S + c0 + t] = sc[t] + prev;
    __syncthreads();
    if (t == 255) carry = prev + sc[255];
    __syncthreads();
  }
}

// ---------------------------------------------------------------------------
// 2) h = ee[x] + emb_rel[clip(td)] + emb_abs[clip(abs_t)]  -> bf16 ; mask=x!=0
// ---------------------------------------------------------------------------
__global__ __launch_bounds__(256) void embed_kernel(
    const int* __restrict__ x, const float* __restrict__ td,
    const float* __restrict__ abs_t, const float* __restrict__ emb_event,
    const float* __restrict__ emb_rel, const float* __restrict__ emb_abs,
    __bf16* __restrict__ h, int* __restrict__ mask) {
  const int i = blockIdx.x, t = threadIdx.x;
  const int xi = x[i];
  int rb = (int)td[i];    rb = rb < 0 ? 0 : (rb > TBINS - 1 ? TBINS - 1 : rb);
  int ab = (int)abs_t[i]; ab = ab < 0 ? 0 : (ab > TBINS - 1 ? TBINS - 1 : ab);
  if (t == 0) mask[i] = (xi != 0) ? 1 : 0;
  const float* er = emb_rel + (size_t)rb * DIMC;
  const float* ea = emb_abs + (size_t)ab * DIMC;
  const float* ee = emb_event + (size_t)xi * DIMC;
  #pragma unroll
  for (int p = 0; p < 2; ++p) {
    int d = t + p * 256;
    float val = er[d] + ea[d] + ((xi != 0) ? ee[d] : 0.0f);  // padding row zeroed
    h[(size_t)i * DIMC + d] = (__bf16)val;
  }
}

// ---------------------------------------------------------------------------
// 3) GEMM: Y[M,N] = A[M,K](bf16) @ W[K,N](f32->bf16) + bias, Y bf16 or f32.
//    Block tile 64x64, K step 64. A staged row-major (async b128), W staged
//    TRANSPOSED (BsT[n][k]) so both fragment reads are K-contiguous b128s.
// ---------------------------------------------------------------------------
#define GT 64
#define GP 72
template <bool OUT_BF16>
__global__ __launch_bounds__(256) void gemm_bias_kernel(
    const __bf16* __restrict__ A, const float* __restrict__ W,
    const float* __restrict__ bias, void* __restrict__ Yv,
    int M, int N, int K) {
  __shared__ __bf16 As[GT * GP];   // As[m][k]
  __shared__ __bf16 BsT[GT * GP];  // BsT[n][k]
  const int t = threadIdx.x;
  const int wave = t >> 5, lane = t & 31;
  const int row0 = blockIdx.x * GT;
  const int col0 = blockIdx.y * GT;
  const int wm  = (wave & 3) * 16;      // M tile
  const int wn0 = (wave >> 2) * 32;     // covers N tiles wn0, wn0+16

  v8f acc0 = {}, acc1 = {};

  for (int k0 = 0; k0 < K; k0 += GT) {
    // stage A tile 64x64 bf16 (async, 16B per lane)
    #pragma unroll
    for (int p = 0; p < 2; ++p) {
      int idx = t + p * 256;
      int r = idx >> 3, c = (idx & 7) * 8;
      async_copy_b128(&As[r * GP + c],
                      A + (size_t)(row0 + r) * K + k0 + c);
    }
    // stage W tile 64x64 f32 -> bf16, transposed into BsT[n][k]
    #pragma unroll
    for (int p = 0; p < 4; ++p) {
      int idx = t + p * 256;
      int r = idx >> 4, c = (idx & 15) * 4;        // r = k-row, c = n-col
      const float4 w4 =
          *reinterpret_cast<const float4*>(W + (size_t)(k0 + r) * N + col0 + c);
      BsT[(c + 0) * GP + r] = (__bf16)w4.x;
      BsT[(c + 1) * GP + r] = (__bf16)w4.y;
      BsT[(c + 2) * GP + r] = (__bf16)w4.z;
      BsT[(c + 3) * GP + r] = (__bf16)w4.w;
    }
    if (k0 + GT < K) {  // pull next tiles toward L2 (global_prefetch_b8)
      __builtin_prefetch(A + (size_t)(row0 + (t & 63)) * K + k0 + GT, 0, 0);
      __builtin_prefetch(W + (size_t)(k0 + GT + (t & 63)) * N + col0, 0, 0);
    }
    async_copy_fence();
    __syncthreads();

    #pragma unroll
    for (int kk = 0; kk < GT; kk += 32) {
      v16bf a, b0, b1;
      const int m  = wm + (lane & 15);
      const int n0 = wn0 + (lane & 15);
      #pragma unroll
      for (int j = 0; j < 16; ++j) {
        const int k = kk + kmap(j, lane);
        a[j]  = As [m * GP + k];
        b0[j] = BsT[n0 * GP + k];
        b1[j] = BsT[(n0 + 16) * GP + k];
      }
      acc0 = wmma_bf16(a, b0, acc0);
      acc1 = wmma_bf16(a, b1, acc1);
    }
    __syncthreads();
  }

  // C/D layout: lane<16 -> N=lane, M=r ; lane>=16 -> N=lane-16, M=8+r
  const int n_ = lane & 15;
  const int mb = (lane >> 4) * 8;
  #pragma unroll
  for (int r = 0; r < 8; ++r) {
    const int m = wm + mb + r;
    const int gn0 = col0 + wn0 + n_;
    const float y0 = acc0[r] + bias[gn0];
    const float y1 = acc1[r] + bias[gn0 + 16];
    const size_t o0 = (size_t)(row0 + m) * N + gn0;
    if constexpr (OUT_BF16) {
      ((__bf16*)Yv)[o0]      = (__bf16)y0;
      ((__bf16*)Yv)[o0 + 16] = (__bf16)y1;
    } else {
      ((float*)Yv)[o0]      = y0;
      ((float*)Yv)[o0 + 16] = y1;
    }
  }
}

// ---------------------------------------------------------------------------
// 4) attention: one block per (sequence-of-64, head). Generic for intra
//    (512 bins) and inter (8 batch rows of 64 bins). D stride = 512.
//    scores = QK^T/8 + temb-bias (masked, else -1e9); softmax; ctx = P @ V.
//    Qs/Ks row-major (fragments K-contiguous); Vs stored TRANSPOSED so the
//    ctx B-fragment is also contiguous.
// ---------------------------------------------------------------------------
__global__ __launch_bounds__(256) void attn_kernel(
    const __bf16* __restrict__ Q, const __bf16* __restrict__ Kx,
    const __bf16* __restrict__ V, const float* __restrict__ td,
    const int* __restrict__ mask, const float* __restrict__ temb,
    __bf16* __restrict__ ctx) {
  __shared__ __bf16 Qs[64 * GP];   // Qs[m][d] then P[m][key]
  __shared__ __bf16 Ks[64 * GP];   // Ks[key][d]
  __shared__ __bf16 VsT[64 * GP];  // VsT[d][key]
  __shared__ float  Sc[64 * 68];
  __shared__ float  sbias[64];
  __shared__ int    smask[64];

  const int seq = blockIdx.x, head = blockIdx.y;
  const int t = threadIdx.x, wave = t >> 5, lane = t & 31;
  const size_t base = (size_t)seq * 64;     // token base
  const int coff = head * 64;               // column (head) offset in D

  #pragma unroll
  for (int p = 0; p < 2; ++p) {
    int idx = t + p * 256;
    int r = idx >> 3, c = (idx & 7) * 8;
    size_t go = (base + r) * DIMC + coff + c;
    async_copy_b128(&Qs[r * GP + c], Q + go);
    async_copy_b128(&Ks[r * GP + c], Kx + go);
    // V transposed through registers (r = key, c..c+7 = d)
    const uint4 vv = *reinterpret_cast<const uint4*>(V + go);
    const __bf16* ve = reinterpret_cast<const __bf16*>(&vv);
    #pragma unroll
    for (int i = 0; i < 8; ++i) VsT[(c + i) * GP + r] = ve[i];
  }
  if (t < 64) {
    const size_t key = base + t;
    int tb = (int)td[key];
    tb = tb < 0 ? 0 : (tb > TBINS - 1 ? TBINS - 1 : tb);
    sbias[t] = temb[(size_t)tb * N_HEADS + head];
    smask[t] = mask[key];
  }
  async_copy_fence();
  __syncthreads();

  // ---- scores = Q K^T : 16 tiles of 16x16, 2 per wave, K(dk)=64 ----
  #pragma unroll
  for (int u = 0; u < 2; ++u) {
    const int tid = wave * 2 + u;
    const int tm = (tid >> 2) * 16, tn = (tid & 3) * 16;
    v8f acc = {};
    #pragma unroll
    for (int kk = 0; kk < 64; kk += 32) {
      v16bf a, b;
      const int mq = tm + (lane & 15);
      const int nk = tn + (lane & 15);
      #pragma unroll
      for (int j = 0; j < 16; ++j) {
        const int k = kk + kmap(j, lane);
        a[j] = Qs[mq * GP + k];
        b[j] = Ks[nk * GP + k];   // B(k=d, n=key) = K[key][d]
      }
      acc = wmma_bf16(a, b, acc);
    }
    const int n_ = tn + (lane & 15);
    const int mb = tm + (lane >> 4) * 8;
    const float bia = sbias[n_];       // hoisted: unconditional LDS loads,
    const bool  ok  = smask[n_] != 0;  // select becomes v_cndmask (no branch)
    #pragma unroll
    for (int r = 0; r < 8; ++r) {
      const float sv = ok ? (acc[r] * 0.125f + bia) : -1e9f;
      Sc[(mb + r) * 68 + n_] = sv;
    }
  }
  __syncthreads();

  // ---- softmax rows; attn probs (bf16) overwrite Qs ----
  if (t < 64) {
    float mx = -3.4e38f;
    #pragma unroll 4
    for (int n = 0; n < 64; ++n) mx = fmaxf(mx, Sc[t * 68 + n]);
    float s = 0.0f;
    #pragma unroll 4
    for (int n = 0; n < 64; ++n) {
      float e = __expf(Sc[t * 68 + n] - mx);
      Sc[t * 68 + n] = e;
      s += e;
    }
    const float inv = 1.0f / s;
    #pragma unroll 4
    for (int n = 0; n < 64; ++n) Qs[t * GP + n] = (__bf16)(Sc[t * 68 + n] * inv);
  }
  __syncthreads();

  // ---- ctx = P @ V ----
  #pragma unroll
  for (int u = 0; u < 2; ++u) {
    const int tid = wave * 2 + u;
    const int tm = (tid >> 2) * 16, tn = (tid & 3) * 16;
    v8f acc = {};
    #pragma unroll
    for (int kk = 0; kk < 64; kk += 32) {
      v16bf a, b;
      const int mq = tm + (lane & 15);
      const int nd = tn + (lane & 15);
      #pragma unroll
      for (int j = 0; j < 16; ++j) {
        const int k = kk + kmap(j, lane);
        a[j] = Qs[mq * GP + k];           // attn probs P[m][key]
        b[j] = VsT[nd * GP + k];          // B(k=key, n=d) = VsT[d][key]
      }
      acc = wmma_bf16(a, b, acc);
    }
    const int n_ = tn + (lane & 15);
    const int mb = tm + (lane >> 4) * 8;
    #pragma unroll
    for (int r = 0; r < 8; ++r)
      ctx[(base + mb + r) * DIMC + coff + n_] = (__bf16)acc[r];
  }
}

// ---------------------------------------------------------------------------
// 5) LayerNorm in place, D=512, one block per row
// ---------------------------------------------------------------------------
__global__ __launch_bounds__(256) void layernorm_kernel(
    float* __restrict__ X, const float* __restrict__ g,
    const float* __restrict__ bt) {
  __shared__ float red[256];
  __shared__ float sMean, sInv;
  const int row = blockIdx.x, t = threadIdx.x;
  float* xr = X + (size_t)row * DIMC;
  const float x0 = xr[t], x1 = xr[t + 256];
  red[t] = x0 + x1;
  __syncthreads();
  for (int off = 128; off > 0; off >>= 1) {
    if (t < off) red[t] += red[t + off];
    __syncthreads();
  }
  if (t == 0) sMean = red[0] * (1.0f / 512.0f);
  __syncthreads();
  const float mean = sMean;
  const float d0 = x0 - mean, d1 = x1 - mean;
  red[t] = d0 * d0 + d1 * d1;
  __syncthreads();
  for (int off = 128; off > 0; off >>= 1) {
    if (t < off) red[t] += red[t + off];
    __syncthreads();
  }
  if (t == 0) sInv = rsqrtf(red[0] * (1.0f / 512.0f) + 1e-5f);
  __syncthreads();
  const float inv = sInv;
  xr[t]       = d0 * inv * g[t] + bt[t];
  xr[t + 256] = d1 * inv * g[t + 256] + bt[t + 256];
}

// ---------------------------------------------------------------------------
// 6) masked mean-pool per bin + bin_mask + bin_abs(max masked abs_t)
// ---------------------------------------------------------------------------
__global__ __launch_bounds__(256) void pool_kernel(
    const float* __restrict__ hi, const int* __restrict__ mask,
    const float* __restrict__ abs_t, __bf16* __restrict__ bin_repr,
    int* __restrict__ bin_mask, float* __restrict__ bin_abs) {
  __shared__ float sden;
  const int bin = blockIdx.x, t = threadIdx.x;
  if (t == 0) {
    int cnt = 0; float mx = 0.0f;
    for (int s = 0; s < 64; ++s) {
      const size_t tok = (size_t)bin * 64 + s;
      if (mask[tok]) { cnt++; mx = fmaxf(mx, abs_t[tok]); }
    }
    bin_mask[bin] = (cnt > 0) ? 1 : 0;
    bin_abs[bin]  = mx;
    sden = fmaxf((float)cnt, 1.0f);
  }
  __syncthreads();
  const float inv = 1.0f / sden;
  #pragma unroll
  for (int p = 0; p < 2; ++p) {
    const int d = t + p * 256;
    float s = 0.0f;
    for (int s_ = 0; s_ < 64; ++s_) {
      const size_t tok = (size_t)bin * 64 + s_;
      if (mask[tok]) s += hi[tok * DIMC + d];
    }
    bin_repr[(size_t)bin * DIMC + d] = (__bf16)(s * inv);
  }
}

// ---------------------------------------------------------------------------
// 7) bin_rel = [0, max(diff(bin_abs),0)] per batch (NB=64)
// ---------------------------------------------------------------------------
__global__ __launch_bounds__(256) void binrel_kernel(
    const float* __restrict__ bin_abs, float* __restrict__ bin_rel, int n) {
  const int i = blockIdx.x * 256 + threadIdx.x;
  if (i >= n) return;
  const int nb = i & 63;
  bin_rel[i] = (nb == 0) ? 0.0f : fmaxf(bin_abs[i] - bin_abs[i - 1], 0.0f);
}

// ---------------------------------------------------------------------------
// 8) head: last valid bin -> relu(last@w1+b1)@w2+b2 -> sigmoid
// ---------------------------------------------------------------------------
__global__ __launch_bounds__(256) void head_kernel(
    const float* __restrict__ h_inter, const int* __restrict__ bin_mask,
    const float* __restrict__ w1, const float* __restrict__ b1,
    const float* __restrict__ w2, const float* __restrict__ b2,
    float* __restrict__ out) {
  __shared__ int slast;
  __shared__ float red[256];
  const int b = blockIdx.x, t = threadIdx.x;
  if (t == 0) {
    int c = 0;
    for (int nb = 0; nb < 64; ++nb) c += bin_mask[b * 64 + nb];
    int li = c - 1; if (li < 0) li = 0;
    slast = li;
  }
  __syncthreads();
  const float* last = h_inter + (size_t)(b * 64 + slast) * DIMC;
  float hsum = b1[t];
  #pragma unroll 4
  for (int kk = 0; kk < DIMC; ++kk) hsum += last[kk] * w1[(size_t)kk * 256 + t];
  hsum = fmaxf(hsum, 0.0f);
  red[t] = hsum * w2[t];
  __syncthreads();
  for (int off = 128; off > 0; off >>= 1) {
    if (t < off) red[t] += red[t + off];
    __syncthreads();
  }
  if (t == 0) out[b] = 1.0f / (1.0f + __expf(-(red[0] + b2[0])));
}

// ---------------------------------------------------------------------------
// launch
// ---------------------------------------------------------------------------
extern "C" void kernel_launch(void* const* d_in, const int* in_sizes, int n_in,
                              void* d_out, int out_size, void* d_ws,
                              size_t ws_size, hipStream_t stream) {
  constexpr size_t B = 8, S = 4096, BS = B * S, NBIN = 512;  // B*NB
  const int*   x         = (const int*)  d_in[0];
  const float* td        = (const float*)d_in[1];
  const float* emb_event = (const float*)d_in[2];
  const float* emb_rel   = (const float*)d_in[3];
  const float* emb_abs   = (const float*)d_in[4];
  const float* iq_w = (const float*)d_in[5];  const float* iq_b = (const float*)d_in[6];
  const float* ik_w = (const float*)d_in[7];  const float* ik_b = (const float*)d_in[8];
  const float* iv_w = (const float*)d_in[9];  const float* iv_b = (const float*)d_in[10];
  const float* io_w = (const float*)d_in[11]; const float* io_b = (const float*)d_in[12];
  const float* eq_w = (const float*)d_in[13]; const float* eq_b = (const float*)d_in[14];
  const float* ek_w = (const float*)d_in[15]; const float* ek_b = (const float*)d_in[16];
  const float* ev_w = (const float*)d_in[17]; const float* ev_b = (const float*)d_in[18];
  const float* eo_w = (const float*)d_in[19]; const float* eo_b = (const float*)d_in[20];
  const float* i_temb = (const float*)d_in[21];
  const float* e_temb = (const float*)d_in[22];
  const float* i_g  = (const float*)d_in[23]; const float* i_bt = (const float*)d_in[24];
  const float* e_g  = (const float*)d_in[25]; const float* e_bt = (const float*)d_in[26];
  const float* w1 = (const float*)d_in[27]; const float* b1 = (const float*)d_in[28];
  const float* w2 = (const float*)d_in[29]; const float* b2 = (const float*)d_in[30];

  // workspace layout (all chunks 256B-aligned sizes)
  constexpr size_t OFF_ABS  = 0;
  constexpr size_t OFF_MASK = OFF_ABS  + BS * 4;
  constexpr size_t OFF_H    = OFF_MASK + BS * 4;
  constexpr size_t OFF_Q    = OFF_H    + BS * DIMC * 2;
  constexpr size_t OFF_K    = OFF_Q    + BS * DIMC * 2;
  constexpr size_t OFF_V    = OFF_K    + BS * DIMC * 2;
  constexpr size_t OFF_CTX  = OFF_V    + BS * DIMC * 2;
  constexpr size_t OFF_AO   = OFF_CTX  + BS * DIMC * 2;
  constexpr size_t OFF_BR   = OFF_AO   + BS * DIMC * 4;
  constexpr size_t OFF_BM   = OFF_BR   + NBIN * DIMC * 2;
  constexpr size_t OFF_BA   = OFF_BM   + NBIN * 4;
  constexpr size_t OFF_BREL = OFF_BA   + NBIN * 4;
  constexpr size_t OFF_QE   = OFF_BREL + NBIN * 4;
  constexpr size_t OFF_KE   = OFF_QE   + NBIN * DIMC * 2;
  constexpr size_t OFF_VE   = OFF_KE   + NBIN * DIMC * 2;
  constexpr size_t OFF_CTXE = OFF_VE   + NBIN * DIMC * 2;
  constexpr size_t OFF_AOE  = OFF_CTXE + NBIN * DIMC * 2;

  char* ws = (char*)d_ws;
  float*  abs_t = (float*) (ws + OFF_ABS);
  int*    maskb = (int*)   (ws + OFF_MASK);
  __bf16* hb    = (__bf16*)(ws + OFF_H);
  __bf16* qb    = (__bf16*)(ws + OFF_Q);
  __bf16* kb    = (__bf16*)(ws + OFF_K);
  __bf16* vb    = (__bf16*)(ws + OFF_V);
  __bf16* ctxb  = (__bf16*)(ws + OFF_CTX);
  float*  aob   = (float*) (ws + OFF_AO);
  __bf16* brb   = (__bf16*)(ws + OFF_BR);
  int*    bmb   = (int*)   (ws + OFF_BM);
  float*  bab   = (float*) (ws + OFF_BA);
  float*  brelb = (float*) (ws + OFF_BREL);
  __bf16* qeb   = (__bf16*)(ws + OFF_QE);
  __bf16* keb   = (__bf16*)(ws + OFF_KE);
  __bf16* veb   = (__bf16*)(ws + OFF_VE);
  __bf16* ctxeb = (__bf16*)(ws + OFF_CTXE);
  float*  aoeb  = (float*) (ws + OFF_AOE);

  // --- stage 0: time features + embeddings ---
  cumsum_kernel<<<B, 256, 0, stream>>>(td, abs_t, (int)S);
  embed_kernel<<<BS, 256, 0, stream>>>(x, td, abs_t, emb_event, emb_rel,
                                       emb_abs, hb, maskb);

  // --- intra-bin attention ---
  const dim3 gI((unsigned)(BS / 64), 512 / 64);  // (512, 8) block tiles
  gemm_bias_kernel<true ><<<gI, 256, 0, stream>>>(hb, iq_w, iq_b, qb, (int)BS, DIMC, DIMC);
  gemm_bias_kernel<true ><<<gI, 256, 0, stream>>>(hb, ik_w, ik_b, kb, (int)BS, DIMC, DIMC);
  gemm_bias_kernel<true ><<<gI, 256, 0, stream>>>(hb, iv_w, iv_b, vb, (int)BS, DIMC, DIMC);
  attn_kernel<<<dim3(NBIN, N_HEADS), 256, 0, stream>>>(qb, kb, vb, td, maskb,
                                                       i_temb, ctxb);
  gemm_bias_kernel<false><<<gI, 256, 0, stream>>>(ctxb, io_w, io_b, aob, (int)BS, DIMC, DIMC);
  layernorm_kernel<<<BS, 256, 0, stream>>>(aob, i_g, i_bt);

  // --- pool to bins ---
  pool_kernel<<<NBIN, 256, 0, stream>>>(aob, maskb, abs_t, brb, bmb, bab);
  binrel_kernel<<<2, 256, 0, stream>>>(bab, brelb, (int)NBIN);

  // --- inter-bin attention (M = 512) ---
  const dim3 gE((unsigned)(NBIN / 64), 512 / 64);  // (8, 8)
  gemm_bias_kernel<true ><<<gE, 256, 0, stream>>>(brb, eq_w, eq_b, qeb, (int)NBIN, DIMC, DIMC);
  gemm_bias_kernel<true ><<<gE, 256, 0, stream>>>(brb, ek_w, ek_b, keb, (int)NBIN, DIMC, DIMC);
  gemm_bias_kernel<true ><<<gE, 256, 0, stream>>>(brb, ev_w, ev_b, veb, (int)NBIN, DIMC, DIMC);
  attn_kernel<<<dim3(B, N_HEADS), 256, 0, stream>>>(qeb, keb, veb, brelb, bmb,
                                                    e_temb, ctxeb);
  gemm_bias_kernel<false><<<gE, 256, 0, stream>>>(ctxeb, eo_w, eo_b, aoeb, (int)NBIN, DIMC, DIMC);
  layernorm_kernel<<<NBIN, 256, 0, stream>>>(aoeb, e_g, e_bt);

  // --- head ---
  head_kernel<<<B, 256, 0, stream>>>(aoeb, bmb, w1, b1, w2, b2, (float*)d_out);
}
